// proposal_attention_model_44865228374114
// MI455X (gfx1250) — matrix operand order repaired and checked by reading, not compiled, not run on hardware
//
#include <hip/hip_runtime.h>
#include <hip/hip_bf16.h>
#include <stdint.h>

#define N_TOT 8192
#define P_DIM 49
#define F_DIM 256
#define E_DIM 256
#define H_DIM 256
#define S_SEG 128

#define NB    5               // n-values per fused block
#define RPAD  256             // padded row count (NB*49 = 245 -> 256)
#define FS    264             // padded LDS row stride (bf16 elems) to avoid bank conflicts

typedef __attribute__((ext_vector_type(16))) __bf16 v16bf;
typedef __attribute__((ext_vector_type(8)))  float  v8f;
typedef uint32_t u32x4 __attribute__((ext_vector_type(4)));
typedef uint32_t u32x8 __attribute__((ext_vector_type(8)));

union FragU { v16bf v; __bf16 e[16]; uint32_t u[8]; uint4 q[2]; };

// ---------------- Kernel 0: W1 (f32) -> W1b (bf16) in workspace ----------------
__global__ void w1bf_kernel(const float* __restrict__ W1, __bf16* __restrict__ W1b)
{
  int idx = blockIdx.x * 256 + threadIdx.x;       // float4 index, 16384 total
  float4 v = ((const float4*)W1)[idx];
  __bf16* d = W1b + (size_t)idx * 4;
  d[0] = (__bf16)v.x; d[1] = (__bf16)v.y; d[2] = (__bf16)v.z; d[3] = (__bf16)v.w;
}

// ---------------- Kernel 1: x2 = tanh(h @ W2^T + b2) ----------------
__global__ void x2_kernel(const float* __restrict__ h,   // [N,H]
                          const float* __restrict__ W2,  // [E,H]
                          const float* __restrict__ b2,  // [E]
                          float* __restrict__ x2)        // [N,E]
{
  const int wave = threadIdx.x >> 5;
  const int lane = threadIdx.x & 31;
  const int hi   = lane >> 4;
  const int l16  = lane & 15;
  const int row0 = blockIdx.x * 16;

  v16bf afrag[8];
  {
    const float* rowp = h + (size_t)(row0 + l16) * H_DIM;
    const int koff = hi * 8;
#pragma unroll
    for (int kt = 0; kt < 8; ++kt) {
      FragU f;
#pragma unroll
      for (int j = 0; j < 8; ++j) {
        int k = kt * 32 + 2 * koff + 2 * j;
        f.e[2 * j]     = (__bf16)rowp[k];
        f.e[2 * j + 1] = (__bf16)rowp[k + 1];
      }
      afrag[kt] = f.v;
    }
  }

#pragma unroll
  for (int t = 0; t < 2; ++t) {
    const int et    = wave * 2 + t;
    const int e_abs = et * 16 + l16;
    const float* brow = W2 + (size_t)e_abs * H_DIM;
    const int khalf = hi * 16;
    v8f c = {};
#pragma unroll
    for (int kt = 0; kt < 8; ++kt) {
      FragU bf;
#pragma unroll
      for (int j = 0; j < 8; ++j) {
        bf.e[2 * j]     = (__bf16)brow[kt * 32 + khalf + 2 * j];
        bf.e[2 * j + 1] = (__bf16)brow[kt * 32 + khalf + 2 * j + 1];
      }
      c = __builtin_amdgcn_wmma_f32_16x16x32_bf16(false, afrag[kt], false, bf.v,
                                                  (short)0, c, false, false);
    }
    const float bias = b2[e_abs];
#pragma unroll
    for (int i = 0; i < 8; ++i) {
      int m = row0 + i + 8 * hi;
      x2[(size_t)m * E_DIM + e_abs] = tanhf(c[i] + bias);
    }
  }
}

// ------- Kernel 2: fused x1-GEMM + tanh + score dot + row-sum of x -------
// W1b (bf16) is staged into LDS by the Tensor Data Mover with per-row padding
// (pad_interval=6 -> every 128 dwords = one 256-elem bf16 row, pad_amount=3 ->
// 4 dwords = 16B), reproducing the FS=264 bank-conflict-free stride in LDS.
__global__ void fused_kernel(const float* __restrict__ x,    // [N,P,F]
                             const __bf16* __restrict__ W1b, // [E,F] bf16
                             const float* __restrict__ b1,   // [E]
                             const float* __restrict__ x2,   // [N,E]
                             float* __restrict__ scores,     // [N]
                             float* __restrict__ xsum)       // [N,F]
{
  extern __shared__ __align__(16) char smem[];
  __bf16* Xs  = (__bf16*)smem;                  // [RPAD][FS]
  __bf16* W1s = Xs + (size_t)RPAD * FS;         // [E_DIM][FS]
  __shared__ float s_x2[NB * E_DIM];
  __shared__ float s_xsum[NB * F_DIM];
  __shared__ float s_score[NB];

  const int tid   = threadIdx.x;
  const int n0    = blockIdx.x * NB;
  const int nCnt  = min(NB, N_TOT - n0);
  const int nrows = nCnt * P_DIM;

  // ---- TDM: DMA W1b -> W1s (wave 0 issues; EXEC ignored; TENSORcnt tracked)
  if (tid < 32) {
    uint64_t ga  = (uint64_t)(uintptr_t)W1b;
    uint32_t lof = (uint32_t)(uintptr_t)W1s;    // LDS byte offset (addr[31:0])
    u32x4 g0;
    g0[0] = 0x1u;                               // count=1, user mode, no gather
    g0[1] = lof;                                // lds_addr
    g0[2] = (uint32_t)ga;                       // global_addr[31:0]
    g0[3] = (uint32_t)((ga >> 32) & 0x1FFFFFFu) | (2u << 30);  // addr[56:32], type=2
    u32x8 g1;
    g1[0] = (1u << 16)      // data_size = 1 (2 bytes)
          | (1u << 20)      // pad_enable
          | (6u << 22)      // pad_interval: 128 dwords (one 256-bf16 row)
          | (3u << 25);     // pad_amount: 4 dwords (16 B) -> stride 264 bf16
    g1[1] = (uint32_t)F_DIM << 16;              // tensor_dim0 = 256 (low 16b)
    g1[2] = (uint32_t)E_DIM << 16;              // tensor_dim1 = 256 (low 16b)
    g1[3] = (uint32_t)F_DIM << 16;              // tile_dim0 = 256
    g1[4] = (uint32_t)E_DIM;                    // tile_dim1 = 256, tile_dim2 = 0
    g1[5] = (uint32_t)F_DIM;                    // tensor_dim0_stride = 256
    g1[6] = 0u;                                 // dim0_stride hi, dim1_stride lo16 (65536->0)
    g1[7] = 1u;                                 // tensor_dim1_stride[47:16] = 1 (=65536)
    asm volatile("tensor_load_to_lds %0, %1" :: "s"(g0), "s"(g1) : "memory");
  }

  for (int i = tid; i < NB * F_DIM; i += 256) s_xsum[i] = 0.f;
  if (tid < NB) s_score[tid] = 0.f;
  __syncthreads();

  // stage this block's x2 rows
  for (int i = tid; i < NB * E_DIM; i += 256) {
    int nn = i / E_DIM;
    s_x2[i] = (nn < nCnt) ? x2[(size_t)(n0 + nn) * E_DIM + (i & (E_DIM - 1))] : 0.f;
  }
  // stage x rows -> LDS bf16, and accumulate per-n row sums (x read exactly once)
  {
    const int chunk = tid & 63;                 // f = 4*chunk .. 4*chunk+3
    for (int r = tid >> 6; r < RPAD; r += 4) {
      float4 v = make_float4(0.f, 0.f, 0.f, 0.f);
      if (r < nrows) {
        v = *(const float4*)(x + ((size_t)n0 * P_DIM + r) * F_DIM + chunk * 4);
        int nn = r / P_DIM;
        float* dx = s_xsum + nn * F_DIM + chunk * 4;
        atomicAdd(dx + 0, v.x); atomicAdd(dx + 1, v.y);
        atomicAdd(dx + 2, v.z); atomicAdd(dx + 3, v.w);
      }
      __bf16* d = Xs + (size_t)r * FS + chunk * 4;
      d[0] = (__bf16)v.x; d[1] = (__bf16)v.y; d[2] = (__bf16)v.z; d[3] = (__bf16)v.w;
    }
  }
  __builtin_amdgcn_s_wait_tensorcnt(0);         // TDM complete (no-op for waves 1..7)
  __syncthreads();

  const int wave = tid >> 5;
  const int lane = tid & 31;
  const int hi   = lane >> 4;
  const int l16  = lane & 15;

  for (int mt = wave * 2; mt < wave * 2 + 2; ++mt) {
    if (mt * 16 >= nrows) continue;
    v16bf afrag[8];
    {
      const __bf16* rowp = Xs + (size_t)(mt * 16 + l16) * FS;
      const int koff = hi * 8;
#pragma unroll
      for (int kt = 0; kt < 8; ++kt) {
        FragU f;
        f.q[0] = *(const uint4*)(rowp + kt * 32 + koff);
        f.q[1] = *(const uint4*)(rowp + kt * 32 + 16 + koff);
        afrag[kt] = f.v;
      }
    }
    float racc[8];
#pragma unroll
    for (int i = 0; i < 8; ++i) racc[i] = 0.f;

    for (int et = 0; et < 16; ++et) {
      const int e_abs = et * 16 + l16;
      const __bf16* brow = W1s + (size_t)e_abs * FS;
      const int khalf = hi * 8;
      v8f c = {};
#pragma unroll
      for (int kt = 0; kt < 8; ++kt) {
        FragU bf;
        bf.q[0] = *(const uint4*)(brow + kt * 32 + khalf);
        bf.q[1] = *(const uint4*)(brow + kt * 32 + khalf + 16);
        c = __builtin_amdgcn_wmma_f32_16x16x32_bf16(false, afrag[kt], false, bf.v,
                                                    (short)0, c, false, false);
      }
      const float bias = b1[e_abs];
#pragma unroll
      for (int i = 0; i < 8; ++i) {
        int r  = mt * 16 + i + 8 * hi;
        int nn = r / P_DIM;
        float x2v = (r < nrows) ? s_x2[nn * E_DIM + e_abs] : 0.f;
        racc[i] += x2v * tanhf(c[i] + bias);
      }
    }
#pragma unroll
    for (int i = 0; i < 8; ++i) {
      float v = racc[i];
      v += __shfl_xor(v, 1, 32);
      v += __shfl_xor(v, 2, 32);
      v += __shfl_xor(v, 4, 32);
      v += __shfl_xor(v, 8, 32);
      racc[i] = v;
    }
    if (l16 == 0) {
#pragma unroll
      for (int i = 0; i < 8; ++i) {
        int r = mt * 16 + i + 8 * hi;
        if (r < nrows) atomicAdd(&s_score[r / P_DIM], racc[i]);
      }
    }
  }
  __syncthreads();

  for (int i = tid; i < nCnt * F_DIM; i += 256)
    xsum[(size_t)n0 * F_DIM + i] = s_xsum[i];
  if (tid < nCnt) scores[n0 + tid] = s_score[tid] * (1.0f / (float)P_DIM);
}

// ---------------- Kernel 3: ragged per-segment softmax ----------------
__global__ void softmax_kernel(const float* __restrict__ scores,
                               const int* __restrict__ lens,    // [S]
                               float* __restrict__ attn)        // [N]
{
  __shared__ float red[256];
  __shared__ int s_sl[2];
  const int s = blockIdx.x, tid = threadIdx.x;
  if (tid == 0) {
    int st = 0;
    for (int j = 0; j < s; ++j) st += lens[j];
    s_sl[0] = st; s_sl[1] = lens[s];
  }
  __syncthreads();
  const int start = s_sl[0], len = s_sl[1];

  float m = -3.4e38f;
  for (int i = tid; i < len; i += 256) m = fmaxf(m, scores[start + i]);
  red[tid] = m; __syncthreads();
  for (int off = 128; off > 0; off >>= 1) {
    if (tid < off) red[tid] = fmaxf(red[tid], red[tid + off]);
    __syncthreads();
  }
  m = red[0]; __syncthreads();

  float z = 0.f;
  for (int i = tid; i < len; i += 256) z += expf(scores[start + i] - m);
  red[tid] = z; __syncthreads();
  for (int off = 128; off > 0; off >>= 1) {
    if (tid < off) red[tid] += red[tid + off];
    __syncthreads();
  }
  z = red[0];
  for (int i = tid; i < len; i += 256)
    attn[start + i] = expf(scores[start + i] - m) / z;
}

// ---------------- Kernel 4: out[n,f] = attn[n] * xsum[n,f] ----------------
__global__ void out_kernel(const float* __restrict__ attn,
                           const float* __restrict__ xsum,
                           float* __restrict__ out)
{
  int idx = blockIdx.x * 256 + threadIdx.x;
  out[idx] = attn[idx >> 8] * xsum[idx];
}

extern "C" void kernel_launch(void* const* d_in, const int* in_sizes, int n_in,
                              void* d_out, int out_size, void* d_ws, size_t ws_size,
                              hipStream_t stream) {
  (void)in_sizes; (void)n_in; (void)out_size; (void)ws_size;
  const float* x  = (const float*)d_in[0];
  const float* h  = (const float*)d_in[1];   // [1,N,H] == [N,H]
  const float* W1 = (const float*)d_in[2];
  const float* b1 = (const float*)d_in[3];
  const float* W2 = (const float*)d_in[4];
  const float* b2 = (const float*)d_in[5];
  const int*  lens = (const int*)d_in[6];
  float* out = (float*)d_out;

  char* ws = (char*)d_ws;
  float* xsum   = (float*)ws;                                         // N*F f32
  float* x2     = (float*)(ws + (size_t)N_TOT * F_DIM * 4);           // N*E f32
  float* scores = (float*)(ws + (size_t)N_TOT * (F_DIM + E_DIM) * 4); // N f32
  float* attn   = scores + N_TOT;                                     // N f32
  __bf16* W1b   = (__bf16*)(attn + N_TOT);                            // E*F bf16 (128 KB)

  const size_t lds_bytes = (size_t)(RPAD + E_DIM) * FS * sizeof(__bf16); // 264 KB
  static bool attr_set = false;
  if (!attr_set) {
    hipFuncSetAttribute((const void*)fused_kernel,
                        hipFuncAttributeMaxDynamicSharedMemorySize, (int)lds_bytes);
    attr_set = true;
  }

  w1bf_kernel<<<(E_DIM * F_DIM / 4) / 256, 256, 0, stream>>>(W1, W1b);
  x2_kernel<<<N_TOT / 16, 256, 0, stream>>>(h, W2, b2, x2);
  const int nblk = (N_TOT + NB - 1) / NB;   // 1639
  fused_kernel<<<nblk, 256, lds_bytes, stream>>>(x, W1b, b1, x2, scores, xsum);
  softmax_kernel<<<S_SEG, 256, 0, stream>>>(scores, lens, attn);
  out_kernel<<<(N_TOT * F_DIM) / 256, 256, 0, stream>>>(attn, xsum, out);
}
